// ExplainableNBRNet_61641370632878
// MI455X (gfx1250) — compile-verified, matrix operand order checked by reading, not run on hardware
//
#include <hip/hip_runtime.h>
#include <hip/hip_bf16.h>
#include <math.h>

// ---------------------------------------------------------------------------
// Problem constants (from the reference): B=512, S=50, H=512, N=20000
// ---------------------------------------------------------------------------
#define BB 512
#define SS 50
#define SSP 64      // S padded to a multiple of 16 (zero rows 50..63)
#define HH 512
#define NN 20000
#define NNP 20032   // N padded to a multiple of 64 (zero rows 20000..20031)

typedef __bf16        v16bf __attribute__((ext_vector_type(16)));
typedef float         v8f   __attribute__((ext_vector_type(8)));
typedef float         v4f   __attribute__((ext_vector_type(4)));
typedef unsigned int  v4u   __attribute__((ext_vector_type(4)));

union BF16Frag {
    v16bf v;
    v4u   q[2];
};

// float -> bf16 bits, round-to-nearest-even
__device__ __forceinline__ unsigned int f2bf(float f) {
    unsigned int u = __float_as_uint(f);
    return (u + 0x7FFFu + ((u >> 16) & 1u)) >> 16;
}

// ---------------------------------------------------------------------------
// f32 -> packed bf16 converter with group row-padding (memory bound).
// dst has `groups` groups of dst_rows rows; src has `groups` groups of
// src_rows rows; rows are cols8*8 elements long.  dst rows >= src_rows get
// zeros, so downstream WMMA fragment loads need no guards at all.
// For plain conversion use src_rows == dst_rows == 1, cols8 = total/8.
// ---------------------------------------------------------------------------
__global__ __launch_bounds__(256)
void f32_to_bf16_pad(const float* __restrict__ src,
                     unsigned short* __restrict__ dst,
                     unsigned cols8, unsigned src_rows, unsigned dst_rows,
                     unsigned total8)
{
    const unsigned idx = blockIdx.x * 256 + threadIdx.x;
    if (idx >= total8) return;
    const unsigned col8 = idx % cols8;
    const unsigned rowg = idx / cols8;
    const unsigned g    = rowg / dst_rows;
    const unsigned row  = rowg % dst_rows;
    v4u o = (v4u)(0u);
    if (row < src_rows) {
        const size_t soff = ((size_t)(g * src_rows + row) * cols8 + col8) * 8;
        const v4f f0 = *(const v4f*)(src + soff);
        const v4f f1 = *(const v4f*)(src + soff + 4);
        o.x = f2bf(f0.x) | (f2bf(f0.y) << 16);
        o.y = f2bf(f0.z) | (f2bf(f0.w) << 16);
        o.z = f2bf(f1.x) | (f2bf(f1.y) << 16);
        o.w = f2bf(f1.z) | (f2bf(f1.w) << 16);
    }
    ((v4u*)dst)[idx] = o;
}

// ---------------------------------------------------------------------------
// Load one 16x32 (MxK) bf16 WMMA fragment from a row-major packed-bf16
// matrix.  ISA layout (CDNA5 ISA, 16-bit A 16x32): lane holds row (lane&15);
// halves 0..7 = K kb..kb+7, halves 8..15 = K kb+16..kb+23, kb = 8*(lane>>4).
// B (KxN) is the mirrored layout with lane = column.  Unguarded (sources are
// zero-padded), 32-bit element offsets so the compiler can use the
// SGPR-base + VGPR-offset (GVS) addressing mode.
// ---------------------------------------------------------------------------
__device__ __forceinline__ v16bf load_frag_bf16(
    const unsigned short* __restrict__ base, unsigned row0, unsigned ld,
    unsigned k0)
{
    const unsigned lane = threadIdx.x & 31;
    const unsigned off  = (row0 + (lane & 15)) * ld + k0 + ((lane >> 4) << 3);
    BF16Frag frag;
    frag.q[0] = *(const v4u*)(base + off);        // K = kb+0..7
    frag.q[1] = *(const v4u*)(base + off + 16);   // K = kb+16..23
    return frag.v;
}

__device__ __forceinline__ v8f wmma_bf16(v16bf a, v16bf b, v8f c) {
    // 8 args: (neg_a, A, neg_b, B, c_mod, C, reuse_a, reuse_b)
    return __builtin_amdgcn_wmma_f32_16x16x32_bf16(
        false, a, false, b, (short)0, c, false, false);
}

// ---------------------------------------------------------------------------
// C[m,n] = sum_k A[m,k] * B[n,k] (both packed bf16, row-major over k, both
// fully valid / zero-padded), f32 accumulate.  Block = 256 threads = 8 waves;
// block tile 128(M) x 64(N); wave tile 64(M) x 16(N).
// Np = padded B rows (load side), N = logical columns (store side).
// ACT: 0 = identity, 1 = sigmoid.
// ---------------------------------------------------------------------------
template <int ACT>
__global__ __launch_bounds__(256)
void gemm_bf16_wmma(const unsigned short* __restrict__ A,
                    const unsigned short* __restrict__ B,
                    float* __restrict__ C, int N, int K)
{
    const int wave = threadIdx.x >> 5;
    const int lane = threadIdx.x & 31;
    const unsigned m0 = blockIdx.y * 128 + (wave & 1) * 64;
    const unsigned n0 = blockIdx.x * 64  + (wave >> 1) * 16;

    v8f acc[4] = {};

    for (unsigned k0 = 0; k0 < (unsigned)K; k0 += 32) {
        const v16bf bfrag = load_frag_bf16(B, n0, K, k0);
        #pragma unroll
        for (int i = 0; i < 4; ++i) {
            const v16bf afrag = load_frag_bf16(A, m0 + 16 * i, K, k0);
            acc[i] = wmma_bf16(afrag, bfrag, acc[i]);
        }
    }

    const int n  = n0 + (lane & 15);
    const int mh = (lane >> 4) * 8;
    if (n < N) {
        #pragma unroll
        for (int i = 0; i < 4; ++i) {
            #pragma unroll
            for (int r = 0; r < 8; ++r) {
                const int m = m0 + 16 * i + mh + r;
                float v = acc[i][r];
                if (ACT == 1) v = 1.0f / (1.0f + __expf(-v));
                C[(size_t)m * N + n] = v;
            }
        }
    }
}

// ---------------------------------------------------------------------------
// Fused additive attention, one block per batch element b:
//   T[s,j]    = sum_h all_memory[b,s,h] * U_w[j,h]          (WMMA, S pad 64)
//   score[s]  = V_b + sum_j V_w[j] * tanh(T[s,j] + lm[b,j]) (LDS ds_add_f32)
//   alpha     = softmax(mask ? -1e9 : score)
//   feat[b]   = bf16([ sum_s alpha[s]*all_memory[b,s,:] | last_memory[b,:] ])
// 8 waves each own 64 consecutive j's (4 N-subtiles) x full M=64.
// ---------------------------------------------------------------------------
__global__ __launch_bounds__(256)
void attn_fused(const unsigned short* __restrict__ am_bf,   // [B,64,H] bf16 padded
                const float* __restrict__ all_memory,       // [B,S,H] f32
                const unsigned short* __restrict__ U_bf,    // [H,H] bf16
                const unsigned short* __restrict__ last_bf, // [B,H] bf16
                const float* __restrict__ V_w,
                const float* __restrict__ V_b,
                const unsigned char* __restrict__ mask,
                const float* __restrict__ lm,               // [B,H] f32
                unsigned short* __restrict__ feat)          // [B,2H] bf16
{
    const int b    = blockIdx.x;
    const int tid  = threadIdx.x;
    const int wave = tid >> 5;
    const int lane = tid & 31;
    const unsigned short* amb_bf = am_bf + (size_t)b * SSP * HH;
    const float*          amb_f  = all_memory + (size_t)b * SS * HH;

    __shared__ float s_scores[64];
    __shared__ float s_alpha[64];
    if (tid < 64) s_scores[tid] = 0.0f;
    __syncthreads();

    const unsigned n0 = wave * 64;  // this wave's j-range: [n0, n0+64)

    v8f acc[4][4] = {};
    for (unsigned k0 = 0; k0 < HH; k0 += 32) {
        v16bf a[4];
        #pragma unroll
        for (int i = 0; i < 4; ++i)
            a[i] = load_frag_bf16(amb_bf, 16 * i, HH, k0);
        #pragma unroll
        for (int j = 0; j < 4; ++j) {
            const v16bf bf = load_frag_bf16(U_bf, n0 + 16 * j, HH, k0);
            #pragma unroll
            for (int i = 0; i < 4; ++i)
                acc[i][j] = wmma_bf16(a[i], bf, acc[i][j]);
        }
    }

    // epilogue: per-lane partial of sum_j V[j]*tanh(T[m,j] + lm[b,j])
    float vw[4], lmv[4];
    #pragma unroll
    for (int j = 0; j < 4; ++j) {
        const int n = n0 + 16 * j + (lane & 15);
        vw[j]  = V_w[n];
        lmv[j] = lm[(size_t)b * HH + n];
    }
    const int mh = (lane >> 4) * 8;
    #pragma unroll
    for (int i = 0; i < 4; ++i) {
        #pragma unroll
        for (int r = 0; r < 8; ++r) {
            const int m = 16 * i + mh + r;
            if (m < SS) {
                float s = 0.0f;
                #pragma unroll
                for (int j = 0; j < 4; ++j)
                    s += vw[j] * tanhf(acc[i][j][r] + lmv[j]);
                atomicAdd(&s_scores[m], s);  // ds_add_f32
            }
        }
    }
    __syncthreads();

    // softmax over s (50 elements: serial on one lane is cheap)
    if (tid == 0) {
        const float vb = V_b[0];
        float mx = -3.0e38f;
        for (int s = 0; s < SS; ++s) {
            float sc = s_scores[s] + vb;
            if (mask[b * SS + s]) sc = -1.0e9f;
            s_scores[s] = sc;
            mx = fmaxf(mx, sc);
        }
        float sum = 0.0f;
        for (int s = 0; s < SS; ++s) {
            float e = __expf(s_scores[s] - mx);
            s_alpha[s] = e;
            sum += e;
        }
        const float inv = 1.0f / sum;
        for (int s = 0; s < SS; ++s) s_alpha[s] *= inv;
    }
    __syncthreads();

    // attention readout (f32 accumulate) + concat -> feat[b, 0:2H] (bf16)
    for (int h = tid; h < HH; h += 256) {
        float c = 0.0f;
        for (int s = 0; s < SS; ++s)
            c += s_alpha[s] * amb_f[(size_t)s * HH + h];
        feat[(size_t)b * (2 * HH) + h]      = (unsigned short)f2bf(c);
        feat[(size_t)b * (2 * HH) + HH + h] = last_bf[(size_t)b * HH + h];
    }
}

// sigmoid(-inf) == 0: zero out logits of already-seen items (item > 0)
__global__ void mask_seen(const int* __restrict__ item_seq,
                          float* __restrict__ out)
{
    const int idx = blockIdx.x * blockDim.x + threadIdx.x;
    if (idx >= BB * SS) return;
    const int b    = idx / SS;
    const int item = item_seq[idx];
    if (item > 0) out[(size_t)b * NN + item] = 0.0f;
}

// ---------------------------------------------------------------------------
extern "C" void kernel_launch(void* const* d_in, const int* in_sizes, int n_in,
                              void* d_out, int out_size, void* d_ws, size_t ws_size,
                              hipStream_t stream)
{
    const float*         all_memory  = (const float*)d_in[0];  // [B,S,H]
    const float*         last_memory = (const float*)d_in[1];  // [B,H]
    const int*           item_seq    = (const int*)d_in[2];    // [B,S]
    const unsigned char* mask        = (const unsigned char*)d_in[3]; // [B,S]
    const float*         U_w         = (const float*)d_in[4];  // [H,H]
    const float*         W_w         = (const float*)d_in[5];  // [H,H]
    const float*         V_w         = (const float*)d_in[6];  // [H]
    const float*         V_b         = (const float*)d_in[7];  // [1]
    const float*         E_w         = (const float*)d_in[8];  // [N,2H]
    float*               out         = (float*)d_out;          // [B,N]

    // ---- workspace layout (all 16B aligned) ------------------------------
    char* ws = (char*)d_ws;
    float*          lm      = (float*)ws;           ws += (size_t)BB * HH * 4;   // 1 MB
    unsigned short* feat_bf = (unsigned short*)ws;  ws += (size_t)BB * 2*HH*2;   // 1 MB
    unsigned short* last_bf = (unsigned short*)ws;  ws += (size_t)BB * HH * 2;   // 0.5 MB
    unsigned short* Ww_bf   = (unsigned short*)ws;  ws += (size_t)HH * HH * 2;   // 0.5 MB
    unsigned short* Uw_bf   = (unsigned short*)ws;  ws += (size_t)HH * HH * 2;   // 0.5 MB
    unsigned short* am_bf   = (unsigned short*)ws;  ws += (size_t)BB*SSP*HH*2;   // 33.5 MB
    unsigned short* Ew_bf   = (unsigned short*)ws;  ws += (size_t)NNP * 2*HH*2;  // 41 MB

    // ---- 0) one-pass f32 -> bf16 conversions w/ zero padding -------------
    {
        unsigned int t8;
        t8 = (BB * HH) / 8;     // last_memory (no padding)
        f32_to_bf16_pad<<<(t8 + 255) / 256, 256, 0, stream>>>(
            last_memory, last_bf, t8, 1u, 1u, t8);
        t8 = (HH * HH) / 8;     // W_w, U_w (no padding)
        f32_to_bf16_pad<<<(t8 + 255) / 256, 256, 0, stream>>>(
            W_w, Ww_bf, t8, 1u, 1u, t8);
        f32_to_bf16_pad<<<(t8 + 255) / 256, 256, 0, stream>>>(
            U_w, Uw_bf, t8, 1u, 1u, t8);
        // all_memory: B groups of S=50 rows -> 64 rows (pad zero)
        t8 = (BB * SSP * HH) / 8;
        f32_to_bf16_pad<<<(t8 + 255) / 256, 256, 0, stream>>>(
            all_memory, am_bf, HH / 8, SS, SSP, t8);
        // E_w: 1 group of 20000 rows -> 20032 rows (pad zero)
        t8 = (unsigned int)(((size_t)NNP * 2 * HH) / 8);
        f32_to_bf16_pad<<<(t8 + 255) / 256, 256, 0, stream>>>(
            E_w, Ew_bf, (2 * HH) / 8, NN, NNP, t8);
    }

    // ---- 1) lm = last_memory @ W_w^T --------------------------------------
    {
        dim3 grid(HH / 64, BB / 128);
        gemm_bf16_wmma<0><<<grid, 256, 0, stream>>>(last_bf, Ww_bf, lm,
                                                    HH, HH);
    }
    // ---- 2) fused attention -> feat = [ctx | last_memory] (bf16) ----------
    attn_fused<<<BB, 256, 0, stream>>>(am_bf, all_memory, Uw_bf, last_bf,
                                       V_w, V_b, mask, lm, feat_bf);
    // ---- 3) out = sigmoid(feat @ E_w^T) -----------------------------------
    {
        dim3 grid(NNP / 64, BB / 128);
        gemm_bf16_wmma<1><<<grid, 256, 0, stream>>>(feat_bf, Ew_bf, out,
                                                    NN, 2 * HH);
    }
    // ---- 4) zero seen items (sigmoid(-inf) == 0) --------------------------
    mask_seen<<<(BB * SS + 255) / 256, 256, 0, stream>>>(item_seq, out);
}
